// SparseTensor_77713138253945
// MI455X (gfx1250) — matrix operand order repaired and checked by reading, not compiled, not run on hardware
//
#include <hip/hip_runtime.h>
#include <hip/hip_bf16.h>
#include <stdint.h>

#define BATCH 2
#define EPS_BN 1e-3f

typedef __attribute__((ext_vector_type(16))) _Float16 v16h;
typedef __attribute__((ext_vector_type(8)))  _Float16 v8h;
typedef __attribute__((ext_vector_type(4)))  _Float16 v4h;
typedef __attribute__((ext_vector_type(8)))  float    v8f;

// ---------------------------------------------------------------- utilities
__global__ void zero_u32_kernel(unsigned* __restrict__ p, long long n) {
  long long i  = (long long)blockIdx.x * blockDim.x + threadIdx.x;
  long long st = (long long)gridDim.x * blockDim.x;
  for (; i < n; i += st) p[i] = 0u;
}

// scatter into halo-padded grid (D+2,H+2,W+2,4)
__global__ void scatter_kernel(const float* __restrict__ feats, const int* __restrict__ coords,
                               int nv, _Float16* __restrict__ grid, float* __restrict__ mask,
                               int D, int H, int W) {
  int i = blockIdx.x * blockDim.x + threadIdx.x;
  if (i >= nv) return;
  int bi = coords[4*i+0], zi = coords[4*i+1], yi = coords[4*i+2], xi = coords[4*i+3];
  long long midx = (((long long)bi * D + zi) * H + yi) * (long long)W + xi;
  mask[midx] = 1.0f;
  long long pidx = (((long long)bi * (D+2) + zi + 1) * (H+2) + yi + 1) * (long long)(W+2)
                   + xi + 1;
  #pragma unroll
  for (int c = 0; c < 4; ++c) grid[pidx*4 + c] = (_Float16)feats[i*4 + c];
}

// weights f32 [k][n] -> f16 transposed zero-padded [n][kpad]
__global__ void w_f16t_kernel(const float* __restrict__ w, _Float16* __restrict__ o,
                              int Ktot, int Kpad, int Cout) {
  int i = blockIdx.x * blockDim.x + threadIdx.x;
  int n = Cout * Kpad;
  if (i >= n) return;
  int col = i / Kpad, k = i - col * Kpad;
  o[i] = (k < Ktot) ? (_Float16)w[k * Cout + col] : (_Float16)0.f;
}

__global__ void mask_down_kernel(const float* __restrict__ mp, float* __restrict__ mc,
                                 int Di, int Hi, int Wi, int Do, int Ho, int Wo) {
  long long Mtot = (long long)BATCH * Do * Ho * Wo;
  long long i = (long long)blockIdx.x * blockDim.x + threadIdx.x;
  if (i >= Mtot) return;
  long long t = i;
  int xo = (int)(t % Wo); t /= Wo;
  int yo = (int)(t % Ho); t /= Ho;
  int zo = (int)(t % Do); int nb = (int)(t / Do);
  float m = 0.f;
  const float* base = mp + (long long)nb * Di * Hi * Wi;
  for (int dz = 0; dz < 3; ++dz)
    for (int dy = 0; dy < 3; ++dy)
      for (int dx = 0; dx < 3; ++dx) {
        int iz = 2*zo + dz - 1, iy = 2*yo + dy - 1, ix = 2*xo + dx - 1;
        if ((unsigned)iz < (unsigned)Di && (unsigned)iy < (unsigned)Hi &&
            (unsigned)ix < (unsigned)Wi)
          if (base[((long long)iz * Hi + iy) * Wi + ix] > 0.f) m = 1.f;
      }
  mc[i] = m;
}

// ------------------------------------------------- implicit-GEMM WMMA conv3d
// Input is halo-padded (DiP=D+2 ...) so NO bounds checks in the K loop.
// Mtot is divisible by 128 for every launch, so no lane/store guards either.
// Addressing: uniform base pointer (SGPR pair) + 32-bit per-lane byte offset
// so A loads use the saddr form with a single v_add per tap.
template <int CIN, int COUT, int STRIDE>
__global__ __launch_bounds__(256)
void conv3d_wmma_kernel(const _Float16* __restrict__ in, const _Float16* __restrict__ wgt,
                        const float* __restrict__ mOut, float* __restrict__ out,
                        int DiP, int HiP, int WiP, int Do, int Ho, int Wo) {
  constexpr int KTOT   = 27 * CIN;
  constexpr int KPAD   = (KTOT + 31) & ~31;
  constexpr int NT     = COUT / 16;
  constexpr int CHUNKS = KPAD / 32;
  __shared__ __align__(64) _Float16 wlds[COUT * KPAD];
  __shared__ int tapbyte_t[32];

  // --- stage transposed weights into LDS with async global->LDS copies
  {
    const int wbytes = COUT * KPAD * 2;               // multiple of 16
    unsigned lbase = (unsigned)(uintptr_t)&wlds[0];   // low 32b of flat = LDS offset
    const unsigned char* g = (const unsigned char*)wgt;
    for (int i = threadIdx.x * 16; i < wbytes; i += blockDim.x * 16) {
      unsigned la = lbase + (unsigned)i;
      unsigned long long ga = (unsigned long long)(uintptr_t)(g + i);
      asm volatile("global_load_async_to_lds_b128 %0, %1, off"
                   :: "v"(la), "v"(ga) : "memory");
    }
    if (threadIdx.x < 32) {            // tap BYTE-offset table (CIN==4 path)
      int tap = threadIdx.x, v = 0;
      if (tap < 27) {
        int dx = tap % 3, dy = (tap / 3) % 3, dz = tap / 9;
        v = ((dz * HiP + dy) * WiP + dx) * CIN * 2;
      }
      tapbyte_t[threadIdx.x] = v;
    }
    asm volatile("s_wait_asynccnt 0x0" ::: "memory");
    __syncthreads();
  }

  const int lane = threadIdx.x & 31;
  const int wv   = threadIdx.x >> 5;
  const int row  = lane & 15;
  const int half = lane >> 4;
  const int m0   = (blockIdx.x * 8 + wv) * 16;

  // per-lane voxel & padded base BYTE offset (computed once; fits in 32 bits)
  int t = m0 + row;
  const int xo = t % Wo; t /= Wo;
  const int yo = t % Ho; t /= Ho;
  const int zo = t % Do; const int nb = t / Do;
  const unsigned laneByte =
      (unsigned)(((((nb * DiP + zo * STRIDE) * HiP + yo * STRIDE) * WiP + xo * STRIDE)
                  * CIN + (CIN >= 8 ? half * 8 : 0)) * 2);
  const char* inB = (const char*)in;

  // hoisted B row pointer (immediate-offset ds loads inside the loop)
  const _Float16* wrow = wlds + row * KPAD + half * 8;

  v8f acc[NT] = {};

  #pragma unroll
  for (int ch = 0; ch < CHUNKS; ++ch) {
    const int k0 = ch * 32;
    // ---- A fragment: element e -> K-in-chunk = (e>=8?16:0)+half*8+(e&7)
    v16h a;
    if constexpr (CIN >= 8) {
      #pragma unroll
      for (int b = 0; b < 2; ++b) {
        const int tap = (k0 + b * 16) / CIN;                 // compile-time
        v8h av = {};
        if (tap < 27) {                                      // folds away
          const int dx = tap % 3, dy = (tap / 3) % 3, dz = tap / 9;
          const int tofB = (((dz * HiP + dy) * WiP + dx) * CIN
                            + (k0 + b * 16) % CIN) * 2;      // wave-uniform bytes
          av = *(const v8h*)(inB + (laneByte + (unsigned)tofB));
        }
        #pragma unroll
        for (int j = 0; j < 8; ++j) a[b * 8 + j] = av[j];
      }
    } else {  // CIN == 4: taps depend on lane-half; use LDS byte-offset table
      #pragma unroll
      for (int b = 0; b < 2; ++b) {
        #pragma unroll
        for (int s2 = 0; s2 < 2; ++s2) {
          const int tap = (k0 + b * 16) / 4 + half * 2 + s2; // runtime (half)
          const unsigned tofB = (unsigned)tapbyte_t[tap & 31]; // tail -> 0 (B is 0)
          v4h av = *(const v4h*)(inB + (laneByte + tofB));
          #pragma unroll
          for (int j = 0; j < 4; ++j) a[b * 8 + s2 * 4 + j] = av[j];
        }
      }
    }
    // ---- B fragments (two 16B ds reads, immediate offsets) + WMMA
    #pragma unroll
    for (int nt = 0; nt < NT; ++nt) {
      v8h b0 = *(const v8h*)(wrow + nt * 16 * KPAD + k0);
      v8h b1 = *(const v8h*)(wrow + nt * 16 * KPAD + k0 + 16);
      v16h bf = __builtin_shufflevector(b0, b1, 0,1,2,3,4,5,6,7,8,9,10,11,12,13,14,15);
      acc[nt] = __builtin_amdgcn_wmma_f32_16x16x32_f16(
          false, a, false, bf, (short)0, acc[nt], false, false);
    }
  }

  // D layout: lane N = lane&15, VGPR r holds M = half*8 + r (no guards needed)
  const int vmBase = m0 + half * 8;
  #pragma unroll
  for (int r = 0; r < 8; ++r) {
    const float mval = mOut[vmBase + r];
    #pragma unroll
    for (int nt = 0; nt < NT; ++nt)
      out[(long long)(vmBase + r) * COUT + nt * 16 + row] = acc[nt][r] * mval;
  }
}

// -------------------------------------------------------------- BN pipeline
template <int C>
__global__ void bn_stats_kernel(const float* __restrict__ y, const float* __restrict__ mask,
                                long long n, float* __restrict__ st) {
  constexpr int CL = (C == 16) ? 4 : 5;
  __shared__ float ssum[C], ssq[C], sc;
  int t = threadIdx.x;
  if (t < C) { ssum[t] = 0.f; ssq[t] = 0.f; }
  if (t == 0) sc = 0.f;
  __syncthreads();
  long long stride = (long long)gridDim.x * blockDim.x;   // multiple of C
  long long i0 = (long long)blockIdx.x * blockDim.x + t;
  int c = (int)(i0 & (C - 1));                            // fixed per thread
  float s = 0.f, q = 0.f, m = 0.f;
  for (long long i = i0; i < n; i += stride) {
    float v = y[i];
    s += v; q += v * v;
    if (c == 0) m += mask[i >> CL];
  }
  atomicAdd(&ssum[c], s);
  atomicAdd(&ssq[c], q);
  if (c == 0) atomicAdd(&sc, m);
  __syncthreads();
  if (t < C) { atomicAdd(&st[t], ssum[t]); atomicAdd(&st[C + t], ssq[t]); }
  if (t == 0) atomicAdd(&st[2 * C], sc);
}

__global__ void bn_finalize_kernel(float* st, const float* __restrict__ g,
                                   const float* __restrict__ b, int C) {
  int c = threadIdx.x;
  if (c >= C) return;
  float cnt = st[2 * C]; cnt = cnt > 1.f ? cnt : 1.f;
  float mean = st[c] / cnt;
  float var  = st[C + c] / cnt - mean * mean;
  float sc   = g[c] * rsqrtf(var + EPS_BN);
  st[c]     = sc;                 // scale
  st[C + c] = b[c] - mean * sc;   // shift
}

// writes activations into halo-padded layout; optional BEV f32 output
template <int C>
__global__ void bn_apply_kernel(const float* __restrict__ y, const float* __restrict__ st,
                                const float* __restrict__ mask, long long n,
                                _Float16* __restrict__ act, float* __restrict__ bev,
                                int Ds, int Hs, int Ws) {
  constexpr int CL = (C == 16) ? 4 : 5;
  long long stride = (long long)gridDim.x * blockDim.x;
  for (long long i = (long long)blockIdx.x * blockDim.x + threadIdx.x; i < n; i += stride) {
    int c = (int)(i & (C - 1));
    long long v = i >> CL;
    float val = st[c] * y[i] + st[C + c];
    val = fmaxf(val, 0.f) * mask[v];
    long long tt = v;
    int w = (int)(tt % Ws); tt /= Ws;
    int h = (int)(tt % Hs); tt /= Hs;
    int z = (int)(tt % Ds); int nbv = (int)(tt / Ds);
    long long pidx = ((((long long)nbv * (Ds+2) + z + 1) * (Hs+2) + h + 1)
                       * (long long)(Ws+2) + w + 1) * C + c;
    act[pidx] = (_Float16)val;
    if (bev)   // BEV: (n, c*D + z, h, w)
      bev[(((long long)nbv * C + c) * Ds + z) * (long long)(Hs * Ws) +
          (long long)h * Ws + w] = val;
  }
}

// ------------------------------------------------------- BN helper (host)
template <int C>
static void run_bn(const float* yconv, const float* mk, const float* g, const float* b,
                   long long nvx, _Float16* actout, float* bev,
                   int Ds, int Hs, int Ws, float* st, hipStream_t stream) {
  zero_u32_kernel<<<1, 128, 0, stream>>>((unsigned*)st, 2 * C + 1);
  long long n = nvx * C;
  bn_stats_kernel<C><<<512, 256, 0, stream>>>(yconv, mk, n, st);
  bn_finalize_kernel<<<1, 32, 0, stream>>>(st, g, b, C);
  long long gb = (n + 255) / 256; if (gb > 4096) gb = 4096;
  bn_apply_kernel<C><<<(int)gb, 256, 0, stream>>>(yconv, st, mk, n, actout, bev,
                                                  Ds, Hs, Ws);
}

// ------------------------------------------------------------------- driver
extern "C" void kernel_launch(void* const* d_in, const int* in_sizes, int n_in,
                              void* d_out, int out_size, void* d_ws, size_t ws_size,
                              hipStream_t stream) {
  (void)n_in; (void)out_size; (void)ws_size;
  const float* feats  = (const float*)d_in[0];
  const int*   coords = (const int*)d_in[1];
  const int NV = in_sizes[0] / 4;

  // params: jax pytree flatten (alphabetical dict keys)
  const float* in_b = (const float*)d_in[3];
  const float* in_g = (const float*)d_in[4];
  const float* in_w = (const float*)d_in[5];
  const float *b_a[6], *b_b[6], *b_sp[6], *g_a[6], *g_b[6], *g_sp[6],
              *w_a[6], *w_b[6], *w_sp[6];
  for (int s = 1; s <= 5; ++s) {
    int base = 6 + (s - 1) * 9;
    b_a[s]  = (const float*)d_in[base + 0];
    b_b[s]  = (const float*)d_in[base + 1];
    b_sp[s] = (const float*)d_in[base + 2];
    g_a[s]  = (const float*)d_in[base + 3];
    g_b[s]  = (const float*)d_in[base + 4];
    g_sp[s] = (const float*)d_in[base + 5];
    w_a[s]  = (const float*)d_in[base + 6];
    w_b[s]  = (const float*)d_in[base + 7];
    w_sp[s] = (const float*)d_in[base + 8];
  }

  static const int Dd[6] = {41, 21, 11, 6, 3, 2};
  static const int Hh[6] = {256, 128, 64, 32, 16, 8};
  static const int Ww[6] = {256, 128, 64, 32, 16, 8};
  long long vox[6], pvox[6];
  for (int s = 0; s < 6; ++s) {
    vox[s]  = (long long)BATCH * Dd[s] * Hh[s] * Ww[s];
    pvox[s] = (long long)BATCH * (Dd[s]+2) * (Hh[s]+2) * (Ww[s]+2);
  }

  char* ws = (char*)d_ws;
  size_t off = 0;
  auto alloc = [&](size_t bytes) {
    size_t o = off; off += (bytes + 255) & ~(size_t)255; return o;
  };
  size_t mask_o[6];
  for (int s = 0; s < 6; ++s) mask_o[s] = alloc(vox[s] * 4);
  size_t grid_o = alloc(pvox[0] * 4 * 2);
  size_t act0_o = alloc(pvox[0] * 16 * 2);
  size_t actA_o[6], actB_o[6];
  for (int s = 1; s <= 5; ++s) { actA_o[s] = alloc(pvox[s] * 32 * 2);
                                 actB_o[s] = alloc(pvox[s] * 32 * 2); }
  size_t conv_o = alloc(vox[0] * 16 * 4);
  size_t wf_o   = alloc((size_t)32 * 864 * 2);   // max COUT*KPAD halfs
  size_t st_o   = alloc(65 * 4);

  float* maskp[6]; for (int s = 0; s < 6; ++s) maskp[s] = (float*)(ws + mask_o[s]);
  _Float16* gridp = (_Float16*)(ws + grid_o);
  _Float16* act0  = (_Float16*)(ws + act0_o);
  _Float16 *actA[6], *actB[6];
  for (int s = 1; s <= 5; ++s) { actA[s] = (_Float16*)(ws + actA_o[s]);
                                 actB[s] = (_Float16*)(ws + actB_o[s]); }
  float*    convp = (float*)(ws + conv_o);
  _Float16* wf    = (_Float16*)(ws + wf_o);
  float*    st    = (float*)(ws + st_o);

  const int TB = 256, ZB = 2048;

  auto wconv = [&](const float* w, int Ktot, int Kpad, int Cout) {
    int n = Cout * Kpad;
    w_f16t_kernel<<<(n + TB - 1) / TB, TB, 0, stream>>>(w, wf, Ktot, Kpad, Cout);
  };

  // ---- zero halo-padded activation buffers + mask0 (every call)
  zero_u32_kernel<<<ZB, TB, 0, stream>>>((unsigned*)gridp, pvox[0] * 2);
  zero_u32_kernel<<<ZB, TB, 0, stream>>>((unsigned*)act0,  pvox[0] * 8);
  for (int s = 1; s <= 5; ++s) {
    zero_u32_kernel<<<ZB, TB, 0, stream>>>((unsigned*)actA[s], pvox[s] * 16);
    zero_u32_kernel<<<ZB, TB, 0, stream>>>((unsigned*)actB[s], pvox[s] * 16);
  }
  zero_u32_kernel<<<ZB, TB, 0, stream>>>((unsigned*)maskp[0], vox[0]);

  // ---- scatter input
  scatter_kernel<<<(NV + TB - 1) / TB, TB, 0, stream>>>(feats, coords, NV, gridp,
                                                        maskp[0], Dd[0], Hh[0], Ww[0]);

  // ---- stage 0: subm (Cin=4 -> 16), Ktot=108 -> Kpad=128
  wconv(in_w, 108, 128, 16);
  conv3d_wmma_kernel<4, 16, 1><<<(int)(vox[0] / 128), TB, 0, stream>>>(
      gridp, wf, maskp[0], convp, Dd[0]+2, Hh[0]+2, Ww[0]+2, Dd[0], Hh[0], Ww[0]);
  run_bn<16>(convp, maskp[0], in_g, in_b, vox[0], act0, nullptr,
             Dd[0], Hh[0], Ww[0], st, stream);

  const _Float16* prev = act0;
  int prevC = 16;
  const long long outoff[6] = {0, 0, 0, 0, 393216, 442368};

  for (int s = 1; s <= 5; ++s) {
    long long Mt = vox[s];
    int cb = (int)(Mt / 128);
    mask_down_kernel<<<(int)((Mt + TB - 1) / TB), TB, 0, stream>>>(
        maskp[s - 1], maskp[s], Dd[s-1], Hh[s-1], Ww[s-1], Dd[s], Hh[s], Ww[s]);

    // spdown (stride 2)
    if (prevC == 16) {
      wconv(w_sp[s], 432, 448, 32);
      conv3d_wmma_kernel<16, 32, 2><<<cb, TB, 0, stream>>>(
          prev, wf, maskp[s], convp, Dd[s-1]+2, Hh[s-1]+2, Ww[s-1]+2,
          Dd[s], Hh[s], Ww[s]);
    } else {
      wconv(w_sp[s], 864, 864, 32);
      conv3d_wmma_kernel<32, 32, 2><<<cb, TB, 0, stream>>>(
          prev, wf, maskp[s], convp, Dd[s-1]+2, Hh[s-1]+2, Ww[s-1]+2,
          Dd[s], Hh[s], Ww[s]);
    }
    run_bn<32>(convp, maskp[s], g_sp[s], b_sp[s], Mt, actA[s], nullptr,
               Dd[s], Hh[s], Ww[s], st, stream);

    // subm a
    wconv(w_a[s], 864, 864, 32);
    conv3d_wmma_kernel<32, 32, 1><<<cb, TB, 0, stream>>>(
        actA[s], wf, maskp[s], convp, Dd[s]+2, Hh[s]+2, Ww[s]+2, Dd[s], Hh[s], Ww[s]);
    run_bn<32>(convp, maskp[s], g_a[s], b_a[s], Mt, actB[s], nullptr,
               Dd[s], Hh[s], Ww[s], st, stream);

    // subm b (+ BEV output for stages 3..5)
    wconv(w_b[s], 864, 864, 32);
    conv3d_wmma_kernel<32, 32, 1><<<cb, TB, 0, stream>>>(
        actB[s], wf, maskp[s], convp, Dd[s]+2, Hh[s]+2, Ww[s]+2, Dd[s], Hh[s], Ww[s]);
    float* bev = (s >= 3) ? ((float*)d_out + outoff[s]) : nullptr;
    run_bn<32>(convp, maskp[s], g_b[s], b_b[s], Mt, actA[s], bev,
               Dd[s], Hh[s], Ww[s], st, stream);

    prev = actA[s];
    prevC = 32;
  }
}
